// QLSTMIDS_87840671137917
// MI455X (gfx1250) — matrix operand order repaired and checked by reading, not compiled, not run on hardware
//
#include <hip/hip_runtime.h>
#include <hip/hip_bf16.h>
#include <math.h>

typedef __attribute__((ext_vector_type(16))) _Float16 v16h;
typedef __attribute__((ext_vector_type(8)))  float    v8f;

#define BQ   4096
#define SQ   512
#define IQ   10
#define HQ   20
#define G4   80      // 4*H
#define NBLK 64      // persistent workgroups (must be co-resident)
#define RPB  64      // batch rows per block (BQ / NBLK)
#define TPB  256     // 8 waves of 32

// ---- workspace layout (32-bit word offsets) ----
#define WS_ABS   0                       // absmax(|x|) as float bits (atomicMax on uint)
#define WS_CNT   1                       // grid-barrier arrive counter
#define WS_GEN   2                       // grid-barrier generation
#define WS_SX    3                       // x scale
#define WS_SWIH  4                       // W_ih scale
#define WS_SWHH  5                       // W_hh scale
#define WS_MAXS  8                       // 2048 per-sync global-max slots
#define WS_NMAX  2048
#define WS_WIHQ  (WS_MAXS + WS_NMAX)     // 800  int codes (as float)
#define WS_WHHQ  (WS_WIHQ + 800)         // 1600 int codes (as float)
#define WS_W1Q   (WS_WHHQ + 1600)        // 1280 fake-quant values
#define WS_W2Q   (WS_W1Q + 1280)         // 2048 fake-quant values
#define WS_W3Q   (WS_W2Q + 2048)         // 160  fake-quant values
#define WS_END   (WS_W3Q + 160)

// ---------------- init: zero control region + max slots ----------------
__global__ void qk_init(unsigned* w) {
  int i = blockIdx.x * blockDim.x + threadIdx.x;
  if (i < WS_MAXS + WS_NMAX) w[i] = 0u;
}

// ---------------- pass 1: per-tensor |x| max ----------------
__global__ void qk_absmax(const float* __restrict__ x, long n, unsigned* slot) {
  __shared__ float red[TPB];
  float m = 0.f;
  for (long i = (long)blockIdx.x * blockDim.x + threadIdx.x; i < n;
       i += (long)gridDim.x * blockDim.x)
    m = fmaxf(m, fabsf(x[i]));
  red[threadIdx.x] = m; __syncthreads();
  for (int s = TPB / 2; s > 0; s >>= 1) {
    if (threadIdx.x < s) red[threadIdx.x] = fmaxf(red[threadIdx.x], red[threadIdx.x + s]);
    __syncthreads();
  }
  if (threadIdx.x == 0) atomicMax(slot, __float_as_uint(red[0]));
}

// ---------------- pass 2: quantize all weights (single block) ----------------
__global__ void qk_prep(const float* __restrict__ Wih, const float* __restrict__ Whh,
                        const float* __restrict__ W1, const float* __restrict__ W2,
                        const float* __restrict__ W3, float* __restrict__ wsf) {
  __shared__ float red[TPB];
  auto bmax = [&](const float* p, int n) -> float {
    float m = 0.f;
    for (int i = threadIdx.x; i < n; i += TPB) m = fmaxf(m, fabsf(p[i]));
    red[threadIdx.x] = m; __syncthreads();
    for (int s = TPB / 2; s > 0; s >>= 1) {
      if (threadIdx.x < s) red[threadIdx.x] = fmaxf(red[threadIdx.x], red[threadIdx.x + s]);
      __syncthreads();
    }
    float r = red[0]; __syncthreads(); return r;
  };
  unsigned* wsu = (unsigned*)wsf;
  float sx = fmaxf(__uint_as_float(wsu[WS_ABS]), 1e-8f) / 127.f;
  float s;
  s = fmaxf(bmax(Wih, G4 * IQ), 1e-8f) / 127.f;
  if (threadIdx.x == 0) { wsf[WS_SX] = sx; wsf[WS_SWIH] = s; }
  for (int i = threadIdx.x; i < G4 * IQ; i += TPB)
    wsf[WS_WIHQ + i] = fminf(fmaxf(rintf(Wih[i] / s), -127.f), 127.f);     // int code
  s = fmaxf(bmax(Whh, G4 * HQ), 1e-8f) / 127.f;
  if (threadIdx.x == 0) wsf[WS_SWHH] = s;
  for (int i = threadIdx.x; i < G4 * HQ; i += TPB)
    wsf[WS_WHHQ + i] = fminf(fmaxf(rintf(Whh[i] / s), -127.f), 127.f);     // int code
  s = fmaxf(bmax(W1, 64 * HQ), 1e-8f) / 127.f;
  for (int i = threadIdx.x; i < 64 * HQ; i += TPB)
    wsf[WS_W1Q + i] = fminf(fmaxf(rintf(W1[i] / s), -127.f), 127.f) * s;   // value
  s = fmaxf(bmax(W2, 32 * 64), 1e-8f) / 127.f;
  for (int i = threadIdx.x; i < 32 * 64; i += TPB)
    wsf[WS_W2Q + i] = fminf(fmaxf(rintf(W2[i] / s), -127.f), 127.f) * s;   // value
  s = fmaxf(bmax(W3, 5 * 32), 1e-8f) / 127.f;
  for (int i = threadIdx.x; i < 5 * 32; i += TPB)
    wsf[WS_W3Q + i] = fminf(fmaxf(rintf(W3[i] / s), -127.f), 127.f) * s;   // value
}

// ---------------- grid-wide barrier (counter+generation) ----------------
__device__ __forceinline__ void gridbar(unsigned* cnt, unsigned* gen) {
  __syncthreads();
  if (threadIdx.x == 0) {
    __threadfence();
    unsigned g = atomicAdd(gen, 0u);
    if (atomicAdd(cnt, 1u) == (unsigned)(NBLK - 1)) {
      atomicExch(cnt, 0u);
      __threadfence();
      atomicAdd(gen, 1u);
    } else {
      while (atomicAdd(gen, 0u) == g) __builtin_amdgcn_s_sleep(2);
    }
    __threadfence();
  }
  __syncthreads();
}

// swizzled slot index for h int-code element (row b, hidden index k):
// matches the 16-bit A-matrix VGPR layout for V_WMMA_*_16X16X32 (ISA 7.12.2)
__device__ __forceinline__ int ha_idx(int b, int k) {
  int tm = b >> 4, m = b & 15;
  int lane = (k >= 8 && k < 16) ? (16 + m) : m;
  int j = (k < 8) ? k : (k - 8);
  return (tm * 32 + lane) * 16 + j;
}

// ---------------- persistent LSTM + MLP ----------------
__global__ void __launch_bounds__(TPB, 1)
qk_lstm_mlp(const float* __restrict__ x, const float* __restrict__ bih,
            const float* __restrict__ bhh, const float* __restrict__ b1,
            const float* __restrict__ b2, const float* __restrict__ b3,
            float* __restrict__ wsf, float* __restrict__ out) {
  // WMMA operands pre-swizzled into register-layout order: one 32B vector
  // LDS read per lane per operand (ds_load_b128 pairs instead of u16 gathers).
  __shared__ __align__(32) _Float16 hAsw[4 * 32 * 16];  // h int codes, A layout
  __shared__ __align__(32) _Float16 Bsw[5 * 32 * 16];   // Whh^T int codes, B layout
  __shared__ float gates[RPB][G4];  // gate accumulators / MLP ping buffer
  __shared__ float cst[RPB][HQ];    // cell state
  __shared__ float xi[RPB][IQ];     // quantized x int codes for current step
  __shared__ float wih[G4][IQ];     // W_ih int codes
  __shared__ float biasL[G4];       // b_ih + b_hh
  __shared__ float mlp[RPB][64];    // MLP pong buffer
  __shared__ float red[TPB];

  unsigned* wsu = (unsigned*)wsf;
  unsigned* cnt = &wsu[WS_CNT];
  unsigned* gen = &wsu[WS_GEN];

  const int tid  = threadIdx.x;
  const int lane = tid & 31;
  const int wv   = tid >> 5;
  const int row0 = blockIdx.x * RPB;

  const float sx   = wsf[WS_SX];
  const float swih = wsf[WS_SWIH];
  const float swhh = wsf[WS_SWHH];
  const float sxw  = sx * swih;

  int sidx = 0;

  auto blockmax = [&](float v) -> float {
    red[tid] = v; __syncthreads();
    for (int k = TPB / 2; k > 0; k >>= 1) {
      if (tid < k) red[tid] = fmaxf(red[tid], red[tid + k]);
      __syncthreads();
    }
    float r = red[0]; __syncthreads(); return r;
  };
  auto globalmax = [&](float v) -> float {
    float bm = blockmax(v);
    if (tid == 0) atomicMax((unsigned*)(wsf + WS_MAXS + sidx), __float_as_uint(bm));
    gridbar(cnt, gen);
    float g = __uint_as_float(*((volatile unsigned*)(wsf + WS_MAXS + sidx)));
    sidx++;
    return g;
  };

  // stage weights (B in WMMA layout), zero state + A buffer (incl. K-padding)
  for (int i = tid; i < G4 * IQ; i += TPB) wih[i / IQ][i % IQ] = wsf[WS_WIHQ + i];
  for (int i = tid; i < 5 * 32 * 16; i += TPB) {
    int tn = i / (32 * 16), l = (i / 16) & 31, j = i & 15;
    int n = tn * 16 + (l & 15);
    int k = ((l < 16) ? 0 : 8) + ((j < 8) ? j : j + 8);
    Bsw[i] = (k < HQ) ? (_Float16)wsf[WS_WHHQ + n * HQ + k] : (_Float16)0.f;
  }
  for (int i = tid; i < G4; i += TPB) biasL[i] = bih[i] + bhh[i];
  for (int i = tid; i < 4 * 32 * 16; i += TPB) hAsw[i] = (_Float16)0.f;
  for (int i = tid; i < RPB * HQ; i += TPB) cst[i / HQ][i % HQ] = 0.f;
  __syncthreads();

  float sh = 0.f;  // h scale (h codes are zero initially, so any scale is exact)
  const v16h* Ap = (const v16h*)hAsw;
  const v16h* Bp = (const v16h*)Bsw;

  for (int s = 0; s < SQ; ++s) {
    // ---- quantize x[:, s, :] slice into LDS int codes ----
    if (tid < RPB) {
      const float* xr = &x[((long)(row0 + tid) * SQ + s) * IQ];
      if (s + 1 < SQ) __builtin_prefetch(&xr[IQ], 0, 1);  // global_prefetch_b8
      #pragma unroll
      for (int i = 0; i < IQ; ++i)
        xi[tid][i] = fminf(fmaxf(rintf(xr[i] / sx), -127.f), 127.f);
    }
    __syncthreads();

    // ---- px = bias + (sx*swih) * xi . Wih^T (K=10, VALU) ----
    #pragma unroll
    for (int q = 0; q < (RPB * G4) / TPB; ++q) {
      int p = tid + q * TPB;
      int b = p / G4, g = p % G4;
      float acc = 0.f;
      #pragma unroll
      for (int i = 0; i < IQ; ++i) acc += xi[b][i] * wih[g][i];
      gates[b][g] = biasL[g] + sxw * acc;
    }
    __syncthreads();

    // ---- recurrent matmul via WMMA f16 (int codes, exact in f32 acc) ----
    // 4 M-tiles x 5 N-tiles = 20 tiles over 8 waves (<=3 per wave)
    const float hw = sh * swhh;
    #pragma unroll
    for (int tt = 0; tt < 3; ++tt) {
      int t = wv + tt * 8;
      if (t < 20) {
        int tm = t & 3, tn = t >> 2;
        v16h A  = Ap[tm * 32 + lane];  // two ds_load_b128 per operand
        v16h Bv = Bp[tn * 32 + lane];
        v8f C = {};
        C = __builtin_amdgcn_wmma_f32_16x16x32_f16(false, A, false, Bv,
                                                   (short)0, C, false, false);
        int roff = tm * 16 + ((lane < 16) ? 0 : 8);
        int ncol = tn * 16 + (lane & 15);
        #pragma unroll
        for (int r = 0; r < 8; ++r)
          gates[roff + r][ncol] += C[r] * hw;
      }
    }
    __syncthreads();

    // ---- gate-accumulator dynamic quant (global max over [B,4H]) ----
    float m = 0.f;
    #pragma unroll
    for (int q = 0; q < (RPB * G4) / TPB; ++q) {
      int p = tid + q * TPB;
      m = fmaxf(m, fabsf(gates[p / G4][p % G4]));
    }
    float sg = fmaxf(globalmax(m), 1e-8f) / 31.f;

    // ---- elementwise gates -> raw cell state ----
    float oreg[5], creg[5];
    float cm = 0.f;
    #pragma unroll
    for (int q = 0; q < (RPB * HQ) / TPB; ++q) {
      int p = tid + q * TPB;
      int b = p / HQ, j = p % HQ;
      float gi = fminf(fmaxf(rintf(gates[b][j]          / sg), -31.f), 31.f) * sg;
      float gf = fminf(fmaxf(rintf(gates[b][HQ + j]     / sg), -31.f), 31.f) * sg;
      float gg = fminf(fmaxf(rintf(gates[b][2 * HQ + j] / sg), -31.f), 31.f) * sg;
      float go = fminf(fmaxf(rintf(gates[b][3 * HQ + j] / sg), -31.f), 31.f) * sg;
      float ii = fminf(fmaxf(rintf((1.f / (1.f + __expf(-gi))) * 63.f), 0.f), 63.f) * (1.f / 63.f);
      float ff = fminf(fmaxf(rintf((1.f / (1.f + __expf(-gf))) * 63.f), 0.f), 63.f) * (1.f / 63.f);
      float oo = fminf(fmaxf(rintf((1.f / (1.f + __expf(-go))) * 63.f), 0.f), 63.f) * (1.f / 63.f);
      float tg = fminf(fmaxf(rintf(tanhf(gg) * 31.f), -31.f), 31.f) * (1.f / 31.f);
      float cr = ff * cst[b][j] + ii * tg;
      creg[q] = cr; oreg[q] = oo;
      cm = fmaxf(cm, fabsf(cr));
    }

    // ---- cell-state dynamic quant (global max) ----
    float sc = fmaxf(globalmax(cm), 1e-8f) / 31.f;
    float hreg[5];
    float hm = 0.f;
    #pragma unroll
    for (int q = 0; q < (RPB * HQ) / TPB; ++q) {
      int p = tid + q * TPB;
      int b = p / HQ, j = p % HQ;
      float cq = fminf(fmaxf(rintf(creg[q] / sc), -31.f), 31.f) * sc;
      cst[b][j] = cq;
      float tc = fminf(fmaxf(rintf(tanhf(cq) * 31.f), -31.f), 31.f) * (1.f / 31.f);
      float hr = oreg[q] * tc;
      hreg[q] = hr;
      hm = fmaxf(hm, fabsf(hr));
    }

    // ---- hidden-state io quant (global max); store codes in A layout ----
    sh = fmaxf(globalmax(hm), 1e-8f) / 127.f;
    #pragma unroll
    for (int q = 0; q < (RPB * HQ) / TPB; ++q) {
      int p = tid + q * TPB;
      int b = p / HQ, j = p % HQ;
      hAsw[ha_idx(b, j)] =
          (_Float16)fminf(fmaxf(rintf(hreg[q] / sh), -127.f), 127.f);
    }
    // next iteration's __syncthreads (xi stage) covers the hAsw writes
  }

  // ================= MLP head =================
  // qrelu(h): 6-bit unsigned dynamic quant, global max over [B,20]
  float m0 = 0.f;
  #pragma unroll
  for (int q = 0; q < (RPB * HQ) / TPB; ++q) {
    int p = tid + q * TPB;
    int b = p / HQ, j = p % HQ;
    float hv = fmaxf((float)hAsw[ha_idx(b, j)] * sh, 0.f);
    gates[b][j] = hv;
    m0 = fmaxf(m0, hv);
  }
  float s0 = fmaxf(globalmax(m0), 1e-8f) / 63.f;
  #pragma unroll
  for (int q = 0; q < (RPB * HQ) / TPB; ++q) {
    int p = tid + q * TPB;
    int b = p / HQ, j = p % HQ;
    gates[b][j] = fminf(fmaxf(rintf(gates[b][j] / s0), 0.f), 63.f) * s0;
  }
  __syncthreads();

  // layer1: 20 -> 64, qrelu
  float m1 = 0.f;
  #pragma unroll
  for (int q = 0; q < (RPB * 64) / TPB; ++q) {
    int p = tid + q * TPB;
    int b = p >> 6, n = p & 63;
    float acc = b1[n];
    #pragma unroll
    for (int k = 0; k < HQ; ++k) acc += gates[b][k] * wsf[WS_W1Q + n * HQ + k];
    acc = fmaxf(acc, 0.f);
    mlp[b][n] = acc;
    m1 = fmaxf(m1, acc);
  }
  float s1 = fmaxf(globalmax(m1), 1e-8f) / 63.f;
  #pragma unroll
  for (int q = 0; q < (RPB * 64) / TPB; ++q) {
    int p = tid + q * TPB;
    int b = p >> 6, n = p & 63;
    mlp[b][n] = fminf(fmaxf(rintf(mlp[b][n] / s1), 0.f), 63.f) * s1;
  }
  __syncthreads();

  // layer2: 64 -> 32, qrelu
  float m2 = 0.f;
  #pragma unroll
  for (int q = 0; q < (RPB * 32) / TPB; ++q) {
    int p = tid + q * TPB;
    int b = p >> 5, n = p & 31;
    float acc = b2[n];
    #pragma unroll
    for (int k = 0; k < 64; ++k) acc += mlp[b][k] * wsf[WS_W2Q + n * 64 + k];
    acc = fmaxf(acc, 0.f);
    gates[b][n] = acc;
    m2 = fmaxf(m2, acc);
  }
  float s2 = fmaxf(globalmax(m2), 1e-8f) / 63.f;
  #pragma unroll
  for (int q = 0; q < (RPB * 32) / TPB; ++q) {
    int p = tid + q * TPB;
    int b = p >> 5, n = p & 31;
    gates[b][n] = fminf(fmaxf(rintf(gates[b][n] / s2), 0.f), 63.f) * s2;
  }
  __syncthreads();

  // layer3: 32 -> 5 (no activation quant)
  for (int p = tid; p < RPB * 5; p += TPB) {
    int b = p / 5, n = p % 5;
    float acc = b3[n];
    #pragma unroll
    for (int k = 0; k < 32; ++k) acc += gates[b][k] * wsf[WS_W3Q + n * 32 + k];
    out[(long)(row0 + b) * 5 + n] = acc;
  }
}

extern "C" void kernel_launch(void* const* d_in, const int* in_sizes, int n_in,
                              void* d_out, int out_size, void* d_ws, size_t ws_size,
                              hipStream_t stream) {
  (void)in_sizes; (void)n_in; (void)out_size; (void)ws_size;
  const float* x   = (const float*)d_in[0];
  const float* Wih = (const float*)d_in[1];
  const float* Whh = (const float*)d_in[2];
  const float* bih = (const float*)d_in[3];
  const float* bhh = (const float*)d_in[4];
  const float* W1  = (const float*)d_in[5];
  const float* b1  = (const float*)d_in[6];
  const float* W2  = (const float*)d_in[7];
  const float* b2  = (const float*)d_in[8];
  const float* W3  = (const float*)d_in[9];
  const float* b3  = (const float*)d_in[10];
  float*    out = (float*)d_out;
  float*    wsf = (float*)d_ws;
  unsigned* wsu = (unsigned*)d_ws;

  qk_init<<<(WS_MAXS + WS_NMAX + TPB - 1) / TPB, TPB, 0, stream>>>(wsu);
  long n = (long)BQ * SQ * IQ;
  qk_absmax<<<1024, TPB, 0, stream>>>(x, n, &wsu[WS_ABS]);
  qk_prep<<<1, TPB, 0, stream>>>(Wih, Whh, W1, W2, W3, wsf);
  qk_lstm_mlp<<<NBLK, TPB, 0, stream>>>(x, bih, bhh, b1, b2, b3, wsf, out);
}
